// Atten_Head_51840255262879
// MI455X (gfx1250) — matrix operand order repaired and checked by reading, not compiled, not run on hardware
//
#include <hip/hip_runtime.h>

// ---------------------------------------------------------------------------
// CDNA5 (gfx1250) attention head: bf16 WMMA (f32 accum), L2-resident staging.
//   Pass 1 (fused x3 over blockIdx.z): C = X(f32) @ W(f32) -> bf16 workspace
//                                      (V written transposed per batch)
//   Pass 2: causal flash attention, fragments loaded straight from L2.
// ---------------------------------------------------------------------------

typedef __attribute__((ext_vector_type(16))) __bf16 v16bf;
typedef __attribute__((ext_vector_type(8)))  float  v8f;

union FragB {
  v16bf v;
  uint4 q[2];
  unsigned short u[16];
};

__device__ __forceinline__ unsigned short f2bf(float f) {
  unsigned int u = __float_as_uint(f);
  u += 0x7FFFu + ((u >> 16) & 1u);  // round-to-nearest-even
  return (unsigned short)(u >> 16);
}

__device__ __forceinline__ unsigned int pack2bf(float a, float b) {
  return (unsigned int)f2bf(a) | ((unsigned int)f2bf(b) << 16);
}

__device__ __forceinline__ v8f vzero8() {
  v8f z;
#pragma unroll
  for (int i = 0; i < 8; ++i) z[i] = 0.0f;
  return z;
}

#define SLEN 2048
#define DDIM 1024
#define QT 32
#define JT 32

// ---------------------------------------------------------------------------
// Fused projection GEMMs: z=0:Q, z=1:K, z=2:V(transposed out).
// Workgroup tile 128x128, 8 waves in 2x4 grid, wave tile 64x32 (4x2 WMMA).
// ---------------------------------------------------------------------------
__global__ __launch_bounds__(256) void proj_gemm(
    const float* __restrict__ Xq, const float* __restrict__ Wq,
    const float* __restrict__ Xk, const float* __restrict__ Wk,
    const float* __restrict__ Xv, const float* __restrict__ Wv,
    unsigned short* __restrict__ Qw, unsigned short* __restrict__ Kw,
    unsigned short* __restrict__ Vtw) {
  const int which = blockIdx.z;
  const float* X = (which == 0) ? Xq : (which == 1) ? Xk : Xv;
  const float* W = (which == 0) ? Wq : (which == 1) ? Wk : Wv;
  unsigned short* Out = (which == 0) ? Qw : (which == 1) ? Kw : Vtw;

  const int tid = threadIdx.x;
  const int lane = tid & 31;
  const int wave = tid >> 5;
  const int wm = wave >> 2;  // 0..1
  const int wn = wave & 3;   // 0..3
  const int tn = blockIdx.x * 128;
  const int tm = blockIdx.y * 128;

  __shared__ __align__(16) unsigned short Xs[128][40];  // 32 k + pad
  __shared__ __align__(16) unsigned short Ws[128][40];  // transposed W tile

  const int klo   = (lane & 16) ? 8 : 0;    // A-frag chunk offset
  const int kb0   = (lane & 16) ? 8 : 0;    // B-frag chunk offset
  const int row16 = lane & 15;
  const int mhalf = (lane >> 4) * 8;

  v8f acc[4][2];
#pragma unroll
  for (int mt = 0; mt < 4; ++mt)
#pragma unroll
    for (int nt = 0; nt < 2; ++nt) acc[mt][nt] = vzero8();

  for (int kb = 0; kb < DDIM; kb += 32) {
    __syncthreads();
    // Stage X tile (128x32): float4 loads, packed bf16x4 ds_store_b64.
#pragma unroll
    for (int i = 0; i < 4; ++i) {
      int u = i * 256 + tid;
      int r = u >> 3, c4 = (u & 7) * 4;
      float4 xv = *(const float4*)&X[(size_t)(tm + r) * DDIM + kb + c4];
      uint2 pk;
      pk.x = pack2bf(xv.x, xv.y);
      pk.y = pack2bf(xv.z, xv.w);
      *(uint2*)&Xs[r][c4] = pk;
    }
    // Stage W tile (32x128) transposed -> Ws[n][k]: 2k x 4n units, packed b32.
#pragma unroll
    for (int i = 0; i < 2; ++i) {
      int u = i * 256 + tid;
      int k = (u >> 5) * 2;        // even k
      int n4 = (u & 31) * 4;       // n chunk
      float4 w0 = *(const float4*)&W[(size_t)(kb + k) * DDIM + tn + n4];
      float4 w1 = *(const float4*)&W[(size_t)(kb + k + 1) * DDIM + tn + n4];
      *(unsigned int*)&Ws[n4 + 0][k] = pack2bf(w0.x, w1.x);
      *(unsigned int*)&Ws[n4 + 1][k] = pack2bf(w0.y, w1.y);
      *(unsigned int*)&Ws[n4 + 2][k] = pack2bf(w0.z, w1.z);
      *(unsigned int*)&Ws[n4 + 3][k] = pack2bf(w0.w, w1.w);
    }
    __syncthreads();

    FragB a[4], b[2];
#pragma unroll
    for (int mt = 0; mt < 4; ++mt) {
      const unsigned short* p = &Xs[wm * 64 + mt * 16 + row16][0];
      a[mt].q[0] = *(const uint4*)(p + klo);
      a[mt].q[1] = *(const uint4*)(p + klo + 16);
    }
#pragma unroll
    for (int nt = 0; nt < 2; ++nt) {
      const unsigned short* p = &Ws[wn * 32 + nt * 16 + row16][0];
      b[nt].q[0] = *(const uint4*)(p + kb0);
      b[nt].q[1] = *(const uint4*)(p + kb0 + 16);
    }
#pragma unroll
    for (int mt = 0; mt < 4; ++mt)
#pragma unroll
      for (int nt = 0; nt < 2; ++nt)
        acc[mt][nt] = __builtin_amdgcn_wmma_f32_16x16x32_bf16(
            false, a[mt].v, false, b[nt].v, (short)0, acc[mt][nt], false, false);
  }

  // Epilogue: uniform branch hoisted out of the store loops.
  if (which == 2) {
#pragma unroll
    for (int mt = 0; mt < 4; ++mt)
#pragma unroll
      for (int nt = 0; nt < 2; ++nt)
#pragma unroll
        for (int i = 0; i < 8; ++i) {
          int mg = tm + wm * 64 + mt * 16 + i + mhalf;  // b*2048 + s
          int ng = tn + wn * 32 + nt * 16 + row16;      // e
          int bb = mg >> 11, s = mg & (SLEN - 1);
          Out[((size_t)bb * DDIM + ng) * SLEN + s] = f2bf(acc[mt][nt][i]);
        }
  } else {
#pragma unroll
    for (int mt = 0; mt < 4; ++mt)
#pragma unroll
      for (int nt = 0; nt < 2; ++nt)
#pragma unroll
        for (int i = 0; i < 8; ++i) {
          int mg = tm + wm * 64 + mt * 16 + i + mhalf;
          int ng = tn + wn * 32 + nt * 16 + row16;
          Out[(size_t)mg * DDIM + ng] = f2bf(acc[mt][nt][i]);
        }
  }
}

// ---------------------------------------------------------------------------
// Causal flash attention. 32 q-rows per WG, 8 waves each own a 128-wide
// e-slice for both score partials and the output accumulator.
// ---------------------------------------------------------------------------
__global__ __launch_bounds__(256) void flash_attn(
    const unsigned short* __restrict__ Qw, const unsigned short* __restrict__ Kw,
    const unsigned short* __restrict__ Vtw, float* __restrict__ Out) {
  const int tid = threadIdx.x;
  const int lane = tid & 31;
  const int wave = tid >> 5;
  const int qb = blockIdx.x * QT;
  const int b = blockIdx.y;

  __shared__ __align__(16) float Sp[8][QT][JT];        // per-wave partials
  __shared__ __align__(16) unsigned short Pl[QT][JT];  // P (bf16)
  __shared__ float alpha_s[QT];
  __shared__ float mrow[QT];
  __shared__ float lrow[QT];

  if (tid < QT) { mrow[tid] = -3.0e38f; lrow[tid] = 0.0f; }

  const int klo   = (lane & 16) ? 8 : 0;
  const int kb0   = (lane & 16) ? 8 : 0;
  const int row16 = lane & 15;
  const int mhalf = (lane >> 4) * 8;

  v8f o[2][8];
#pragma unroll
  for (int mb = 0; mb < 2; ++mb)
#pragma unroll
    for (int nt = 0; nt < 8; ++nt) o[mb][nt] = vzero8();

  const unsigned short* Qbase = Qw + (size_t)b * SLEN * DDIM;
  const unsigned short* Kbase = Kw + (size_t)b * SLEN * DDIM;
  const unsigned short* Vbase = Vtw + (size_t)b * DDIM * SLEN;

  const float scale = 0.022097086912079608f;  // 1/sqrt(2048)
  const int nj = qb / JT + 1;                 // causal: tiles up to diagonal

  for (int j = 0; j < nj; ++j) {
    const int jb = j * JT;

    // ---- Phase A: partial scores over this wave's 128-wide e-slice -------
    v8f sacc[2][2];
#pragma unroll
    for (int mb = 0; mb < 2; ++mb)
#pragma unroll
      for (int nb = 0; nb < 2; ++nb) sacc[mb][nb] = vzero8();

#pragma unroll
    for (int ec = 0; ec < 4; ++ec) {
      const int ebase = wave * 128 + ec * 32;
      FragB a[2], bk[2];
#pragma unroll
      for (int mb = 0; mb < 2; ++mb) {
        const unsigned short* p =
            Qbase + (size_t)(qb + mb * 16 + row16) * DDIM + ebase;
        a[mb].q[0] = *(const uint4*)(p + klo);
        a[mb].q[1] = *(const uint4*)(p + klo + 16);
      }
#pragma unroll
      for (int nb = 0; nb < 2; ++nb) {
        const unsigned short* p =
            Kbase + (size_t)(jb + nb * 16 + row16) * DDIM + ebase;
        bk[nb].q[0] = *(const uint4*)(p + kb0);
        bk[nb].q[1] = *(const uint4*)(p + kb0 + 16);
      }
#pragma unroll
      for (int mb = 0; mb < 2; ++mb)
#pragma unroll
        for (int nb = 0; nb < 2; ++nb)
          sacc[mb][nb] = __builtin_amdgcn_wmma_f32_16x16x32_bf16(
              false, a[mb].v, false, bk[nb].v, (short)0, sacc[mb][nb], false,
              false);
    }
#pragma unroll
    for (int mb = 0; mb < 2; ++mb)
#pragma unroll
      for (int nb = 0; nb < 2; ++nb)
#pragma unroll
        for (int i = 0; i < 8; ++i)
          Sp[wave][mb * 16 + i + mhalf][nb * 16 + row16] = sacc[mb][nb][i];
    __syncthreads();

    // ---- Phase B: deterministic cross-wave reduce + online softmax -------
    if (tid < QT) {
      const int r = tid;
      const int qg = qb + r;
      float mold = mrow[r];
      float mnew = mold;
      float sv[JT];
#pragma unroll
      for (int c4 = 0; c4 < JT / 4; ++c4) {
        float4 a = *(const float4*)&Sp[0][r][c4 * 4];
#pragma unroll
        for (int w = 1; w < 8; ++w) {
          float4 t = *(const float4*)&Sp[w][r][c4 * 4];
          a.x += t.x; a.y += t.y; a.z += t.z; a.w += t.w;
        }
        float tmp[4] = {a.x, a.y, a.z, a.w};
#pragma unroll
        for (int jj = 0; jj < 4; ++jj) {
          int c = c4 * 4 + jj;
          float s = tmp[jj] * scale;
          if (jb + c > qg) s = -3.0e38f;  // causal mask
          sv[c] = s;
          mnew = fmaxf(mnew, s);
        }
      }
      float sum = 0.0f;
#pragma unroll
      for (int c = 0; c < JT; c += 2) {
        float p0 = __expf(sv[c] - mnew);
        float p1 = __expf(sv[c + 1] - mnew);
        sum += p0 + p1;
        *(unsigned int*)&Pl[r][c] = pack2bf(p0, p1);
      }
      float al = __expf(mold - mnew);
      lrow[r] = lrow[r] * al + sum;
      mrow[r] = mnew;
      alpha_s[r] = al;
    }
    __syncthreads();

    // ---- Phase C: rescale O and accumulate P @ V ------------------------
    float al2[2][8];
#pragma unroll
    for (int mb = 0; mb < 2; ++mb)
#pragma unroll
      for (int i = 0; i < 8; ++i) al2[mb][i] = alpha_s[mb * 16 + i + mhalf];

    FragB ap[2];
#pragma unroll
    for (int mb = 0; mb < 2; ++mb) {
      const unsigned short* p = &Pl[mb * 16 + row16][0];
      ap[mb].q[0] = *(const uint4*)(p + klo);
      ap[mb].q[1] = *(const uint4*)(p + klo + 16);
    }
#pragma unroll
    for (int nt = 0; nt < 8; ++nt) {
      const int e = wave * 128 + nt * 16 + row16;
      const unsigned short* p = Vbase + (size_t)e * SLEN + jb;
      FragB bv;
      bv.q[0] = *(const uint4*)(p + kb0);
      bv.q[1] = *(const uint4*)(p + kb0 + 16);
#pragma unroll
      for (int mb = 0; mb < 2; ++mb) {
        v8f t = o[mb][nt];
#pragma unroll
        for (int i = 0; i < 8; ++i) t[i] *= al2[mb][i];
        o[mb][nt] = __builtin_amdgcn_wmma_f32_16x16x32_bf16(
            false, ap[mb].v, false, bv.v, (short)0, t, false, false);
      }
    }
    // No trailing barrier needed: next Sp writes are ordered by Phase A's
    // barrier; Pl/alpha rewritten only after all waves pass it.
  }

  // ---- Epilogue: O / l, f32 out ------------------------------------------
  float li[2][8];
#pragma unroll
  for (int mb = 0; mb < 2; ++mb)
#pragma unroll
    for (int i = 0; i < 8; ++i) li[mb][i] = 1.0f / lrow[mb * 16 + i + mhalf];
#pragma unroll
  for (int mb = 0; mb < 2; ++mb)
#pragma unroll
    for (int nt = 0; nt < 8; ++nt)
#pragma unroll
      for (int i = 0; i < 8; ++i) {
        int qg = qb + mb * 16 + i + mhalf;
        int e = wave * 128 + nt * 16 + row16;
        Out[((size_t)b * SLEN + qg) * DDIM + e] = o[mb][nt][i] * li[mb][i];
      }
}

// ---------------------------------------------------------------------------
extern "C" void kernel_launch(void* const* d_in, const int* in_sizes, int n_in,
                              void* d_out, int out_size, void* d_ws,
                              size_t ws_size, hipStream_t stream) {
  (void)in_sizes; (void)n_in; (void)out_size; (void)ws_size;
  const float* Xk = (const float*)d_in[0];
  const float* Xv = (const float*)d_in[1];
  const float* Xq = (const float*)d_in[2];
  const float* Wk = (const float*)d_in[3];
  const float* Wv = (const float*)d_in[4];
  const float* Wq = (const float*)d_in[5];
  float* out = (float*)d_out;

  const size_t mat = (size_t)4 * SLEN * DDIM;  // 8M bf16 elements each
  unsigned short* Qw = (unsigned short*)d_ws;
  unsigned short* Kw = Qw + mat;
  unsigned short* Vtw = Kw + mat;  // 48 MB total workspace

  dim3 gp(DDIM / 128, (4 * SLEN) / 128, 3), bp(256);
  proj_gemm<<<gp, bp, 0, stream>>>(Xq, Wq, Xk, Wk, Xv, Wv, Qw, Kw, Vtw);

  dim3 ga(SLEN / QT, 4), ba(256);
  flash_attn<<<ga, ba, 0, stream>>>(Qw, Kw, Vtw, out);
}
